// Parking_Trajectory_Planner_4913442587022
// MI455X (gfx1250) — matrix operand order repaired and checked by reading, not compiled, not run on hardware
//
#include <hip/hip_runtime.h>
#include <math.h>

typedef __attribute__((ext_vector_type(16))) _Float16 v16h;
typedef __attribute__((ext_vector_type(8)))  _Float16 v8h;
typedef __attribute__((ext_vector_type(8)))  float    v8f;

namespace {
constexpr int kB   = 64;    // batch
constexpr int kNPT = 2048;  // map points
constexpr int kS   = 4;     // stages
constexpr int kA   = 8;     // anchors per stage
constexpr int kH   = 256;   // hidden
constexpr int kL   = 2;     // lstm layers
constexpr int kLOC = 3;
constexpr int kG   = 128;   // ogm grid 128x128
constexpr int kCOUT = 8;
constexpr int kO   = 64;    // conv output spatial
constexpr int kOO  = kO * kO; // 4096
}

__device__ __forceinline__ float sigmf(float x) { return 1.0f / (1.0f + expf(-x)); }

// ---------------------------------------------------------------------------
// fp32 -> fp16 weight staging (once per launch; weights then live in L2)
// ---------------------------------------------------------------------------
__global__ void cvt_kernel(const float* __restrict__ s, _Float16* __restrict__ d, int n) {
  for (int i = blockIdx.x * blockDim.x + threadIdx.x; i < n; i += gridDim.x * blockDim.x)
    d[i] = (_Float16)s[i];
}

__global__ void zero_kernel(float* __restrict__ p, int n) {
  for (int i = blockIdx.x * blockDim.x + threadIdx.x; i < n; i += gridDim.x * blockDim.x)
    p[i] = 0.0f;
}

// conv weights: all 4 input channels of xin are identical -> pre-sum over CIN.
// conv_w: [S][COUT][4][25] -> wsum: [S][COUT][25]
__global__ void conv_presum_kernel(const float* __restrict__ w, float* __restrict__ wsum) {
  int i = blockIdx.x * blockDim.x + threadIdx.x;
  if (i < kS * kCOUT * 25) {
    int st = i / (kCOUT * 25), r = i % (kCOUT * 25);
    int o = r / 25, k = r % 25;
    float s = 0.0f;
    for (int c = 0; c < 4; ++c) s += w[(((st * kCOUT + o) * 4 + c) * 25) + k];
    wsum[i] = s;
  }
}

__global__ void anchor_init_kernel(const float* __restrict__ inp, float* __restrict__ anchor,
                                   float* __restrict__ varb) {
  int i = blockIdx.x * blockDim.x + threadIdx.x;
  if (i < kB * kLOC) { anchor[i] = inp[i]; varb[i] = 0.01f; }
}

__global__ void hc_init_kernel(_Float16* __restrict__ h16, float* __restrict__ c32, int n) {
  for (int i = blockIdx.x * blockDim.x + threadIdx.x; i < n; i += gridDim.x * blockDim.x) {
    h16[i] = (_Float16)1.0f;
    c32[i] = 1.0f;
  }
}

// write anchor/var state into outputs at step index 0 of a stage
__global__ void stage_out_kernel(const float* __restrict__ anchor, const float* __restrict__ varb,
                                 float* __restrict__ om, float* __restrict__ ov, int st) {
  int b = blockIdx.x, t = threadIdx.x; // block = 3
  int idx = ((b * kS + st) * kA + 0) * kLOC + t;
  om[idx] = anchor[b * kLOC + t];
  ov[idx] = varb[b * kLOC + t];
}

// ---------------------------------------------------------------------------
// OGM: rotate/translate map points into ego frame, scatter 1s into grid
// ---------------------------------------------------------------------------
__global__ void ogm_kernel(const float* __restrict__ map_pts, const float* __restrict__ anchor,
                           float* __restrict__ grid) {
  int b = blockIdx.x;
  float ax = anchor[b * 3 + 0], ay = anchor[b * 3 + 1], th = anchor[b * 3 + 2];
  float c = cosf(th), s = sinf(th);
  float* g = grid + (size_t)b * kG * kG;
  for (int p = threadIdx.x; p < kNPT; p += blockDim.x) {
    float rx = map_pts[p] - ax;
    float ry = map_pts[kNPT + p] - ay;
    float x = c * rx + s * ry;
    float y = -s * rx + c * ry;
    if (fabsf(x) > 12.8f || fabsf(y) > 12.8f) { x = 0.0f; y = 0.0f; }
    int xi = (int)floorf(x * 5.0f + 64.0f); // nearest bin center, 0.2m bins
    int yi = (int)floorf(y * 5.0f + 64.0f);
    xi = xi < 0 ? 0 : (xi > 127 ? 127 : xi);
    yi = yi < 0 ? 0 : (yi > 127 ? 127 : yi);
    g[xi * kG + yi] = 1.0f; // racy identical writes: benign
  }
}

// ---------------------------------------------------------------------------
// conv 5x5 stride2 pad2 (CIN pre-summed), tanh, sum over 8 out-channels.
// grid (b, i), block 64 threads = j.  Output -> f16 [64][4096] (elm input)
// ---------------------------------------------------------------------------
__global__ void conv_kernel(const float* __restrict__ grid, const float* __restrict__ wsum,
                            _Float16* __restrict__ out) {
  __shared__ float w[kCOUT * 25];
  int b = blockIdx.x, i = blockIdx.y, j = threadIdx.x;
  for (int t = j; t < kCOUT * 25; t += 64) w[t] = wsum[t];
  __syncthreads();
  const float* g = grid + (size_t)b * kG * kG;
  float in[25];
#pragma unroll
  for (int kh = 0; kh < 5; ++kh) {
    int x = i * 2 - 2 + kh;
#pragma unroll
    for (int kw = 0; kw < 5; ++kw) {
      int y = j * 2 - 2 + kw;
      in[kh * 5 + kw] = (x >= 0 && x < kG && y >= 0 && y < kG) ? g[x * kG + y] : 0.0f;
    }
  }
  float acc = 0.0f;
  for (int o = 0; o < kCOUT; ++o) {
    float s = 0.0f;
#pragma unroll
    for (int t = 0; t < 25; ++t) s += in[t] * w[o * 25 + t];
    acc += tanhf(s);
  }
  out[(size_t)b * kOO + i * kO + j] = (_Float16)acc;
}

// ---------------------------------------------------------------------------
// ela layer 1: [64,3] @ [256,3]^T + relu  (K=3 -> plain VALU, negligible)
// ---------------------------------------------------------------------------
__global__ void ela1_kernel(const float* __restrict__ anchor, const float* __restrict__ w1,
                            _Float16* __restrict__ out) {
  int b = blockIdx.x, j = threadIdx.x;
  float s = anchor[b * 3 + 0] * w1[j * 3 + 0] + anchor[b * 3 + 1] * w1[j * 3 + 1] +
            anchor[b * 3 + 2] * w1[j * 3 + 2];
  out[b * kH + j] = (_Float16)fmaxf(s, 0.0f);
}

// ---------------------------------------------------------------------------
// WMMA GEMM:  C[64 x N] = A1[64 x K] @ W1[N x K]^T  (+ A2 @ W2^T for LSTM)
// f16 inputs, f32 accumulate via v_wmma_f32_16x16x32_f16. One wave = one
// 16x16 C tile. Optional ReLU; optional f32 and/or f16 outputs (strided).
// Fragment layouts per CDNA5 ISA §7.12.2.
// ---------------------------------------------------------------------------
union HF16 { v16h v; v8h h8[2]; };

__global__ void wmma_gemm_kernel(const _Float16* __restrict__ A1, const _Float16* __restrict__ W1,
                                 const _Float16* __restrict__ A2, const _Float16* __restrict__ W2,
                                 int K, float* __restrict__ Cf, _Float16* __restrict__ Ch,
                                 int ldc, int relu) {
  const int lane = threadIdx.x & 31;
  const int lm = lane & 15;
  const int hi = lane >> 4;
  const int m0 = blockIdx.y << 4;
  const int n0 = blockIdx.x << 4;
  v8f acc = {};
  {
    const _Float16* ar = A1 + (size_t)(m0 + lm) * K;
    const _Float16* wr = W1 + (size_t)(n0 + lm) * K;
    for (int k = 0; k < K; k += 32) {
      HF16 a, b;
      a.h8[0] = *(const v8h*)(ar + k + hi * 8);       // A: K = hi*8 .. +7
      a.h8[1] = *(const v8h*)(ar + k + 16 + hi * 8);  // A: K = 16+hi*8 .. +7
      b.h8[0] = *(const v8h*)(wr + k + hi * 8);       // B: K = hi*8 .. +7
      b.h8[1] = *(const v8h*)(wr + k + 16 + hi * 8);  // B: K = 16+hi*8 .. +7
      acc = __builtin_amdgcn_wmma_f32_16x16x32_f16(false, a.v, false, b.v,
                                                   (short)0, acc, false, false);
    }
  }
  if (A2 != nullptr) {
    const _Float16* ar = A2 + (size_t)(m0 + lm) * K;
    const _Float16* wr = W2 + (size_t)(n0 + lm) * K;
    for (int k = 0; k < K; k += 32) {
      HF16 a, b;
      a.h8[0] = *(const v8h*)(ar + k + hi * 8);
      a.h8[1] = *(const v8h*)(ar + k + 16 + hi * 8);
      b.h8[0] = *(const v8h*)(wr + k + hi * 8);
      b.h8[1] = *(const v8h*)(wr + k + 16 + hi * 8);
      acc = __builtin_amdgcn_wmma_f32_16x16x32_f16(false, a.v, false, b.v,
                                                   (short)0, acc, false, false);
    }
  }
#pragma unroll
  for (int v = 0; v < 8; ++v) { // C/D: m = hi*8+v, n = lane&15
    int m = m0 + hi * 8 + v;
    int n = n0 + lm;
    float val = acc[v];
    if (relu) val = fmaxf(val, 0.0f);
    size_t o = (size_t)m * ldc + n;
    if (Cf) Cf[o] = val;
    if (Ch) Ch[o] = (_Float16)val;
  }
}

// ---------------------------------------------------------------------------
// LSTM gate nonlinearity + state update for one layer
// ---------------------------------------------------------------------------
__global__ void lstm_gates_kernel(const float* __restrict__ g, float* __restrict__ c,
                                  _Float16* __restrict__ h16, _Float16* __restrict__ x16,
                                  float* __restrict__ hlast32) {
  int b = blockIdx.x, j = threadIdx.x;
  const float* gr = g + (size_t)b * (4 * kH);
  float gi = gr[j], gf = gr[kH + j], gg = gr[2 * kH + j], go = gr[3 * kH + j];
  float cv = c[b * kH + j];
  float cn = sigmf(gf) * cv + sigmf(gi) * tanhf(gg);
  float hn = sigmf(go) * tanhf(cn);
  c[b * kH + j] = cn;
  _Float16 hh = (_Float16)hn;
  h16[b * kH + j] = hh;  // recurrent state for this layer
  x16[b * kH + j] = hh;  // input to next layer / residual chain
  hlast32[b * kH + j] = hn;
}

// ---------------------------------------------------------------------------
// LayerNorm(hlast) + ebl, relu -> r16
// ---------------------------------------------------------------------------
__global__ void ln_relu_kernel(const float* __restrict__ x, const float* __restrict__ ebl,
                               _Float16* __restrict__ r16) {
  __shared__ float red[kH];
  int b = blockIdx.x, j = threadIdx.x;
  float v = x[b * kH + j];
  red[j] = v;
  __syncthreads();
  for (int s = kH / 2; s > 0; s >>= 1) {
    if (j < s) red[j] += red[j + s];
    __syncthreads();
  }
  float mu = red[0] * (1.0f / kH);
  __syncthreads();
  float d = v - mu;
  red[j] = d * d;
  __syncthreads();
  for (int s = kH / 2; s > 0; s >>= 1) {
    if (j < s) red[j] += red[j + s];
    __syncthreads();
  }
  float var = red[0] * (1.0f / kH);
  float ln = d * rsqrtf(var + 1e-5f);
  r16[b * kH + j] = (_Float16)fmaxf(ln + ebl[b * kH + j], 0.0f);
}

// ---------------------------------------------------------------------------
// heads: dm2/dv2 (N=3), tanh*dlm + anchor, sigmoid*0.1; update state + outputs
// ---------------------------------------------------------------------------
__global__ void head_kernel(const _Float16* __restrict__ dmb, const _Float16* __restrict__ dvb,
                            const float* __restrict__ dm_w2, const float* __restrict__ dv_w2,
                            float* __restrict__ anchor, float* __restrict__ varb,
                            float* __restrict__ om, float* __restrict__ ov, int st, int step) {
  __shared__ float rm[kH], rv[kH];
  int b = blockIdx.x, t = threadIdx.x;
  rm[t] = (float)dmb[b * kH + t];
  rv[t] = (float)dvb[b * kH + t];
  __syncthreads();
  if (t < kLOC) {
    float sm = 0.0f, sv = 0.0f;
    for (int k = 0; k < kH; ++k) {
      sm += rm[k] * dm_w2[t * kH + k];
      sv += rv[k] * dv_w2[t * kH + k];
    }
    const float dlm = (t == 2) ? 0.5f : 2.0f;
    float an = tanhf(sm) * dlm + anchor[b * kLOC + t];
    float vn = sigmf(sv) * 0.1f;
    anchor[b * kLOC + t] = an;
    varb[b * kLOC + t] = vn;
    int idx = ((b * kS + st) * kA + step) * kLOC + t;
    om[idx] = an;
    ov[idx] = vn;
  }
}

// ---------------------------------------------------------------------------
extern "C" void kernel_launch(void* const* d_in, const int* in_sizes, int n_in,
                              void* d_out, int out_size, void* d_ws, size_t ws_size,
                              hipStream_t stream) {
  (void)in_sizes; (void)n_in; (void)out_size; (void)ws_size;
  const float* inp_start = (const float*)d_in[0];
  const float* map_pts   = (const float*)d_in[1];
  const float* ela_w1    = (const float*)d_in[2];
  const float* ela_w2    = (const float*)d_in[3];
  const float* conv_w    = (const float*)d_in[4];
  const float* elm_w1    = (const float*)d_in[5];
  const float* elm_w2    = (const float*)d_in[6];
  const float* ebl_w1    = (const float*)d_in[7];
  const float* ebl_w2    = (const float*)d_in[8];
  const float* lstm_wih  = (const float*)d_in[9];
  const float* lstm_whh  = (const float*)d_in[10];
  const float* dm_w1     = (const float*)d_in[11];
  const float* dm_w2     = (const float*)d_in[12];
  const float* dv_w1     = (const float*)d_in[13];
  const float* dv_w2     = (const float*)d_in[14];
  float* out_mean = (float*)d_out;
  float* out_var  = out_mean + (size_t)kB * kS * kA * kLOC;

  // bump allocator over workspace (~26 MB total, L2-resident)
  char* base = (char*)d_ws;
  size_t off = 0;
  auto alloc = [&](size_t bytes) -> void* {
    off = (off + 255) & ~(size_t)255;
    void* p = base + off;
    off += bytes;
    return p;
  };
  _Float16* w_ela2 = (_Float16*)alloc(sizeof(_Float16) * kS * kH * kH);
  _Float16* w_elm1 = (_Float16*)alloc(sizeof(_Float16) * kS * kH * kOO);
  _Float16* w_elm2 = (_Float16*)alloc(sizeof(_Float16) * kS * kH * kH);
  _Float16* w_ebl1 = (_Float16*)alloc(sizeof(_Float16) * kS * kH * 2 * kH);
  _Float16* w_ebl2 = (_Float16*)alloc(sizeof(_Float16) * kS * kH * kH);
  _Float16* w_wih  = (_Float16*)alloc(sizeof(_Float16) * kS * kL * 4 * kH * kH);
  _Float16* w_whh  = (_Float16*)alloc(sizeof(_Float16) * kS * kL * 4 * kH * kH);
  _Float16* w_dm1  = (_Float16*)alloc(sizeof(_Float16) * kS * kH * kH);
  _Float16* w_dv1  = (_Float16*)alloc(sizeof(_Float16) * kS * kH * kH);
  float*    wsum   = (float*)alloc(sizeof(float) * kS * kCOUT * 25);
  float*    grid   = (float*)alloc(sizeof(float) * kB * kG * kG);
  _Float16* conv16 = (_Float16*)alloc(sizeof(_Float16) * kB * kOO);
  _Float16* ela1b  = (_Float16*)alloc(sizeof(_Float16) * kB * kH);
  _Float16* tmp16  = (_Float16*)alloc(sizeof(_Float16) * kB * kH);
  _Float16* concat = (_Float16*)alloc(sizeof(_Float16) * kB * 2 * kH);
  _Float16* x16    = (_Float16*)alloc(sizeof(_Float16) * kB * kH);
  _Float16* h16    = (_Float16*)alloc(sizeof(_Float16) * kL * kB * kH);
  _Float16* r16    = (_Float16*)alloc(sizeof(_Float16) * kB * kH);
  _Float16* dmb    = (_Float16*)alloc(sizeof(_Float16) * kB * kH);
  _Float16* dvb    = (_Float16*)alloc(sizeof(_Float16) * kB * kH);
  float*    ebl32  = (float*)alloc(sizeof(float) * kB * kH);
  float*    g32    = (float*)alloc(sizeof(float) * kB * 4 * kH);
  float*    c32    = (float*)alloc(sizeof(float) * kL * kB * kH);
  float*    hl32   = (float*)alloc(sizeof(float) * kB * kH);
  float*    anchor = (float*)alloc(sizeof(float) * kB * kLOC);
  float*    varb   = (float*)alloc(sizeof(float) * kB * kLOC);

  auto cvt = [&](const float* s, _Float16* d, int n) {
    cvt_kernel<<<256, 256, 0, stream>>>(s, d, n);
  };
  cvt(ela_w2, w_ela2, kS * kH * kH);
  cvt(elm_w1, w_elm1, kS * kH * kOO);
  cvt(elm_w2, w_elm2, kS * kH * kH);
  cvt(ebl_w1, w_ebl1, kS * kH * 2 * kH);
  cvt(ebl_w2, w_ebl2, kS * kH * kH);
  cvt(lstm_wih, w_wih, kS * kL * 4 * kH * kH);
  cvt(lstm_whh, w_whh, kS * kL * 4 * kH * kH);
  cvt(dm_w1, w_dm1, kS * kH * kH);
  cvt(dv_w1, w_dv1, kS * kH * kH);
  conv_presum_kernel<<<4, 256, 0, stream>>>(conv_w, wsum);
  anchor_init_kernel<<<1, 256, 0, stream>>>(inp_start, anchor, varb);

  auto gemm = [&](const _Float16* A1, const _Float16* W1, const _Float16* A2, const _Float16* W2,
                  int K, int N, float* Cf, _Float16* Ch, int ldc, int relu) {
    dim3 g(N / 16, kB / 16);
    wmma_gemm_kernel<<<g, 32, 0, stream>>>(A1, W1, A2, W2, K, Cf, Ch, ldc, relu);
  };

  for (int st = 0; st < kS; ++st) {
    hc_init_kernel<<<64, 256, 0, stream>>>(h16, c32, kL * kB * kH);
    stage_out_kernel<<<kB, kLOC, 0, stream>>>(anchor, varb, out_mean, out_var, st);
    for (int step = 0; step < kA - 1; ++step) {
      // ela branch
      ela1_kernel<<<kB, kH, 0, stream>>>(anchor, ela_w1 + (size_t)st * kH * kLOC, ela1b);
      gemm(ela1b, w_ela2 + (size_t)st * kH * kH, nullptr, nullptr,
           kH, kH, nullptr, concat + 0, 2 * kH, 0);
      // OGM -> conv -> elm branch
      zero_kernel<<<512, 256, 0, stream>>>(grid, kB * kG * kG);
      ogm_kernel<<<kB, 256, 0, stream>>>(map_pts, anchor, grid);
      conv_kernel<<<dim3(kB, kO), kO, 0, stream>>>(grid, wsum + st * kCOUT * 25, conv16);
      gemm(conv16, w_elm1 + (size_t)st * kH * kOO, nullptr, nullptr,
           kOO, kH, nullptr, tmp16, kH, 1);
      gemm(tmp16, w_elm2 + (size_t)st * kH * kH, nullptr, nullptr,
           kH, kH, nullptr, concat + kH, 2 * kH, 0);
      // ebl on concat
      gemm(concat, w_ebl1 + (size_t)st * kH * 2 * kH, nullptr, nullptr,
           2 * kH, kH, nullptr, tmp16, kH, 1);
      gemm(tmp16, w_ebl2 + (size_t)st * kH * kH, nullptr, nullptr,
           kH, kH, ebl32, x16, kH, 0);
      // 2-layer LSTM: g = x @ Wih^T + h @ Whh^T (dual-input WMMA GEMM)
      for (int l = 0; l < kL; ++l) {
        const size_t wo = ((size_t)st * kL + l) * 4 * kH * kH;
        gemm(x16, w_wih + wo, h16 + (size_t)l * kB * kH, w_whh + wo,
             kH, 4 * kH, g32, nullptr, 4 * kH, 0);
        lstm_gates_kernel<<<kB, kH, 0, stream>>>(g32, c32 + (size_t)l * kB * kH,
                                                 h16 + (size_t)l * kB * kH, x16, hl32);
      }
      // LN + ebl + relu, then heads
      ln_relu_kernel<<<kB, kH, 0, stream>>>(hl32, ebl32, r16);
      gemm(r16, w_dm1 + (size_t)st * kH * kH, nullptr, nullptr, kH, kH, nullptr, dmb, kH, 1);
      gemm(r16, w_dv1 + (size_t)st * kH * kH, nullptr, nullptr, kH, kH, nullptr, dvb, kH, 1);
      head_kernel<<<kB, kH, 0, stream>>>(dmb, dvb, dm_w2 + (size_t)st * kLOC * kH,
                                         dv_w2 + (size_t)st * kLOC * kH,
                                         anchor, varb, out_mean, out_var, st, step + 1);
    }
  }
}
